// GATOnlyNet_38474317037827
// MI455X (gfx1250) — matrix operand order repaired and checked by reading, not compiled, not run on hardware
//
#include <hip/hip_runtime.h>

// ---------------------------------------------------------------------------
// GAT (3 layers, V=50000, E=1.6M, Fin=H*C=128, H=4, C=32) for gfx1250.
// GEMM uses v_wmma_f32_16x16x32_f16 (f16 in, f32 accumulate). Edge phase is
// L2-resident scatter/gather with f32 atomics (z/agg fit in the 192MB L2).
// ---------------------------------------------------------------------------

#define V_ 50000
#define E_ 1600000
#define F_ 128          // Fin == HEADS*COUT == 128 for every layer
#define H_ 4
#define C_ 32
#define NEG_SLOPE 0.2f

typedef __attribute__((ext_vector_type(16))) _Float16 v16h;
typedef __attribute__((ext_vector_type(8)))  _Float16 v8h;
typedef __attribute__((ext_vector_type(8)))  float    v8f;

// Order-preserving encoding of f32 into u32 so atomicMax(uint) == float max.
__device__ __forceinline__ unsigned f32_ord(float f) {
    unsigned u = __float_as_uint(f);
    return (u & 0x80000000u) ? ~u : (u | 0x80000000u);
}
__device__ __forceinline__ float ord_f32(unsigned u) {
    u = (u & 0x80000000u) ? (u ^ 0x80000000u) : ~u;
    return __uint_as_float(u);
}

// --------------------------- weight f32 -> f16 -----------------------------
__global__ void convert_w_f16(const float* __restrict__ w, _Float16* __restrict__ wh, int n) {
    int i = blockIdx.x * blockDim.x + threadIdx.x;
    if (i < n) wh[i] = (_Float16)w[i];
}

// --------------------------- z = h @ W^T via WMMA --------------------------
// Block = 256 threads = 8 waves. Block handles 16 node rows; wave w handles
// output channels [16w, 16w+16). K = 128 -> 4 wmma_f32_16x16x32_f16 per wave.
__global__ __launch_bounds__(256) void gat_gemm_wmma(const float* __restrict__ hin,
                                                     const _Float16* __restrict__ wh,
                                                     float* __restrict__ z) {
    __shared__ __align__(32) _Float16 sh[16 * F_];   // 16x128 f16 activation tile
    const int t  = threadIdx.x;
    const int m0 = blockIdx.x * 16;                  // V = 3125*16 exactly, no tail

    for (int i = t; i < 16 * F_; i += 256)
        sh[i] = (_Float16)hin[m0 * F_ + i];
    __syncthreads();

    const int wave = t >> 5;
    const int lane = t & 31;
    const int hi   = lane >> 4;      // half-wave select (0 or 1)
    const int l15  = lane & 15;
    const int n0   = wave * 16;      // 8 waves cover N = 128

    v8f acc = {};
    #pragma unroll
    for (int kt = 0; kt < 4; ++kt) {
        // A (16x32 f16): lane row = l15; lanes 0-15 hold K {0..15}, 16-31 {16..31}
        const int ka = kt * 32 + 16 * hi;
        const v8h alo = *(const v8h*)(&sh[l15 * F_ + ka]);        // ds_load_b128
        const v8h ahi = *(const v8h*)(&sh[l15 * F_ + ka + 8]);    // ds_load_b128
        const v16h a = __builtin_shufflevector(alo, ahi,
                          0,1,2,3,4,5,6,7,8,9,10,11,12,13,14,15);
        // B (32x16 f16): lane column N = n0+l15; K = kt*32 + 16*hi + j (contiguous)
        const int kb = kt * 32 + 16 * hi;
        const v16h b = *(const v16h*)(&wh[(n0 + l15) * F_ + kb]); // B[k][n] = W[n][k]
        acc = __builtin_amdgcn_wmma_f32_16x16x32_f16(
                  false, a, false, b, (short)0, acc, false, false);
    }
    // C/D layout: element r -> M = r + 8*hi, N = l15
    #pragma unroll
    for (int r = 0; r < 8; ++r)
        z[(m0 + r + 8 * hi) * F_ + n0 + l15] = acc[r];
}

// --------------------- per-node attention scores ---------------------------
__global__ void gat_scores(const float* __restrict__ z,
                           const float* __restrict__ asrc, const float* __restrict__ adst,
                           float* __restrict__ ssrc, float* __restrict__ sdst) {
    int i = blockIdx.x * blockDim.x + threadIdx.x;
    if (i >= V_ * H_) return;
    int v = i / H_, h = i % H_;
    float a = 0.f, b = 0.f;
    #pragma unroll 8
    for (int c = 0; c < C_; ++c) {
        float zc = z[v * F_ + h * C_ + c];
        a += zc * asrc[h * C_ + c];
        b += zc * adst[h * C_ + c];
    }
    ssrc[i] = a;
    sdst[i] = b;
}

// ----------------------- per-layer state init ------------------------------
__global__ void gat_init(float* __restrict__ agg, unsigned* __restrict__ mord,
                         float* __restrict__ denom) {
    int i = blockIdx.x * blockDim.x + threadIdx.x;
    if (i < V_ * F_) agg[i] = 0.f;
    if (i < V_ * H_) { mord[i] = f32_ord(-1e9f); denom[i] = 0.f; }
}

// ------------- edge pass 1: leaky-relu score + segment max -----------------
__global__ void gat_edge_max(const int* __restrict__ src, const int* __restrict__ dst,
                             const float* __restrict__ ssrc, const float* __restrict__ sdst,
                             float* __restrict__ ebuf, unsigned* __restrict__ mord) {
    int i = blockIdx.x * blockDim.x + threadIdx.x;
    if (i >= E_ * H_) return;
    int e = i >> 2, h = i & 3;
    int s = src[e], d = dst[e];
    float eh = ssrc[s * H_ + h] + sdst[d * H_ + h];
    eh = (eh > 0.f) ? eh : NEG_SLOPE * eh;
    ebuf[i] = eh;
    atomicMax(&mord[d * H_ + h], f32_ord(eh));
}

// ------------- edge pass 2: exp(e - m) + segment sum -----------------------
__global__ void gat_edge_exp(const int* __restrict__ dst, float* __restrict__ ebuf,
                             const unsigned* __restrict__ mord, float* __restrict__ denom) {
    int i = blockIdx.x * blockDim.x + threadIdx.x;
    if (i >= E_ * H_) return;
    int e = i >> 2, h = i & 3;
    int d = dst[e];
    float ex = __expf(ebuf[i] - ord_f32(mord[d * H_ + h]));
    ebuf[i] = ex;
    atomicAdd(&denom[d * H_ + h], ex);
}

// ------------- edge pass 3: alpha * z[src] scattered into agg[dst] ---------
// 2 edges per 256-thread block; 128 consecutive lanes cover one edge's
// 128 channels -> coalesced z gather and coalesced f32 L2 atomics.
__global__ __launch_bounds__(256) void gat_edge_agg(const int* __restrict__ src,
                                                    const int* __restrict__ dst,
                                                    const float* __restrict__ ebuf,
                                                    const float* __restrict__ denom,
                                                    const float* __restrict__ z,
                                                    float* __restrict__ agg) {
    int t = threadIdx.x;
    int e = blockIdx.x * 2 + (t >> 7);
    int j = t & 127;             // channel 0..127
    int h = j >> 5;              // head
    int s = src[e], d = dst[e];
    float alpha = ebuf[e * H_ + h] / (denom[d * H_ + h] + 1e-9f);
    atomicAdd(&agg[d * F_ + j], z[s * F_ + j] * alpha);
}

// --------------------------- ELU activation --------------------------------
__global__ void gat_elu(const float* __restrict__ agg, float* __restrict__ hout) {
    int i = blockIdx.x * blockDim.x + threadIdx.x;
    if (i >= V_ * F_) return;
    float a = agg[i];
    hout[i] = (a > 0.f) ? a : expm1f(a);
}

// --------------------------- output head -----------------------------------
__global__ void gat_head(const float* __restrict__ h, const float* __restrict__ w,
                         const float* __restrict__ b, float* __restrict__ out) {
    int v = blockIdx.x * blockDim.x + threadIdx.x;
    if (v >= V_) return;
    const float4* hv = (const float4*)(h + v * F_);
    const float4* wv = (const float4*)w;
    float acc = b[0];
    #pragma unroll
    for (int j = 0; j < F_ / 4; ++j) {
        float4 a = hv[j], c = wv[j];
        acc += a.x * c.x + a.y * c.y + a.z * c.z + a.w * c.w;
    }
    out[v] = acc;
}

// ---------------------------------------------------------------------------
extern "C" void kernel_launch(void* const* d_in, const int* in_sizes, int n_in,
                              void* d_out, int out_size, void* d_ws, size_t ws_size,
                              hipStream_t stream) {
    const float* x    = (const float*)d_in[0];
    const int*   ei   = (const int*)d_in[1];   // JAX default x64-off -> int32
    const float* Ws   = (const float*)d_in[2];
    const float* asrc = (const float*)d_in[3];
    const float* adst = (const float*)d_in[4];
    const float* hw   = (const float*)d_in[5];
    const float* hb   = (const float*)d_in[6];
    float* out = (float*)d_out;
    const int* src = ei;
    const int* dst = ei + E_;

    // workspace carve-up (256B aligned slices)
    char* p = (char*)d_ws;
    auto take = [&p](size_t bytes) { char* r = p; p += (bytes + 255) & ~size_t(255); return r; };
    float*    z     = (float*)take(sizeof(float) * V_ * F_);
    float*    hbuf  = (float*)take(sizeof(float) * V_ * F_);
    float*    agg   = (float*)take(sizeof(float) * V_ * F_);
    float*    ebuf  = (float*)take(sizeof(float) * (size_t)E_ * H_);
    float*    ssrc  = (float*)take(sizeof(float) * V_ * H_);
    float*    sdst  = (float*)take(sizeof(float) * V_ * H_);
    unsigned* mord  = (unsigned*)take(sizeof(unsigned) * V_ * H_);
    float*    denom = (float*)take(sizeof(float) * V_ * H_);
    _Float16* wh    = (_Float16*)take(sizeof(_Float16) * 3 * F_ * F_);

    convert_w_f16<<<(3 * F_ * F_ + 255) / 256, 256, 0, stream>>>(Ws, wh, 3 * F_ * F_);

    for (int li = 0; li < 3; ++li) {
        const float* hin = (li == 0) ? x : hbuf;
        gat_gemm_wmma<<<V_ / 16, 256, 0, stream>>>(hin, wh + li * F_ * F_, z);
        gat_scores<<<(V_ * H_ + 255) / 256, 256, 0, stream>>>(
            z, asrc + li * H_ * C_, adst + li * H_ * C_, ssrc, sdst);
        gat_init<<<(V_ * F_ + 255) / 256, 256, 0, stream>>>(agg, mord, denom);
        gat_edge_max<<<(E_ * H_ + 255) / 256, 256, 0, stream>>>(src, dst, ssrc, sdst, ebuf, mord);
        gat_edge_exp<<<(E_ * H_ + 255) / 256, 256, 0, stream>>>(dst, ebuf, mord, denom);
        gat_edge_agg<<<E_ / 2, 256, 0, stream>>>(src, dst, ebuf, denom, z, agg);
        gat_elu<<<(V_ * F_ + 255) / 256, 256, 0, stream>>>(agg, hbuf);
    }
    gat_head<<<(V_ + 255) / 256, 256, 0, stream>>>(hbuf, hw, hb, out);
}
